// TFLiteLinearSelfAttention_72730976190922
// MI455X (gfx1250) — matrix operand order repaired and checked by reading, not compiled, not run on hardware
//
#include <hip/hip_runtime.h>

// ---------------------------------------------------------------------------
// TFLite linear self-attention, MI455X (gfx1250, wave32).
// GEMMs run on V_WMMA_F32_16X16X32_BF16 (fp32 accumulate); softmax / context
// vector in fp32. Intermediate k/v kept in bf16 in workspace (~135 MB).
// Weights pre-converted to bf16 once; fp32->bf16 packing is a 3-op sequence
// (bias-add 0x8000 + V_PERM_B32).
// ---------------------------------------------------------------------------

#define CCH   512          // channels (K dim of both GEMMs)
#define PN    4096         // P*N spatial columns per batch
#define NB    16           // batch
#define BT_M  128          // block tile M
#define BT_N  128          // block tile N
#define KC    32           // K chunk (== one bf16 WMMA K)
#define LSTR  36           // padded LDS row stride (bf16 elems)

typedef __attribute__((ext_vector_type(16))) __bf16 v16bf;
typedef __attribute__((ext_vector_type(8)))  float  v8f;

union BF16Frag { unsigned u[8]; v16bf v; };

__device__ __forceinline__ float bf2f(unsigned short h) {
  return __uint_as_float(((unsigned)h) << 16);
}
// Round-half-up to bf16 (0.5 ulp, unbiased for zero-centered data); the pair
// merge is a single V_PERM_B32.
__device__ __forceinline__ unsigned pack2(float lo, float hi) {
  const unsigned ul = __float_as_uint(lo) + 0x8000u;
  const unsigned uh = __float_as_uint(hi) + 0x8000u;
  return __builtin_amdgcn_perm(uh, ul, 0x07060302u);
}
__device__ __forceinline__ unsigned short f2bf(float f) {
  return (unsigned short)((__float_as_uint(f) + 0x8000u) >> 16);
}

// One K=32 chunk of the 128x128 block tile. Wave tile: 32(M) x 64(N).
// Ws: bf16 [128 m][32 k] stride LSTR. Xs: bf16 [128 n][32 k] stride LSTR
// (transposed, so every B-fragment dword is a contiguous K-pair).
__device__ __forceinline__ void mma_chunk(const unsigned short* Ws,
                                          const unsigned short* Xs,
                                          int waveM, int waveN, int lane,
                                          v8f acc[2][4]) {
  const int lm = lane & 15;
  const int hi = lane >> 4;
  BF16Frag afr[2];
#pragma unroll
  for (int a = 0; a < 2; ++a) {
    // 16-bit A layout: VGPR j<4 -> K{2j,2j+1}+8*hi ; j>=4 -> K{16+2(j-4)..}+8*hi
    const unsigned short* ap = Ws + (waveM * 32 + a * 16 + lm) * LSTR + hi * 8;
#pragma unroll
    for (int j = 0; j < 4; ++j) afr[a].u[j]     = *(const unsigned*)(ap + 2 * j);
#pragma unroll
    for (int j = 0; j < 4; ++j) afr[a].u[4 + j] = *(const unsigned*)(ap + 16 + 2 * j);
  }
#pragma unroll
  for (int s = 0; s < 4; ++s) {
    // 16-bit B layout: VGPR j<4 -> K{2j,2j+1}+8*hi ; j>=4 -> K{16+2(j-4)..}+8*hi, column n = lane&15
    BF16Frag bfr;
    const unsigned short* bp = Xs + (waveN * 64 + s * 16 + lm) * LSTR + hi * 8;
#pragma unroll
    for (int j = 0; j < 4; ++j) bfr.u[j] = *(const unsigned*)(bp + 2 * j);
#pragma unroll
    for (int j = 0; j < 4; ++j) bfr.u[4 + j] = *(const unsigned*)(bp + 16 + 2 * j);
    acc[0][s] = __builtin_amdgcn_wmma_f32_16x16x32_bf16(
        false, afr[0].v, false, bfr.v, (short)0, acc[0][s], false, false);
    acc[1][s] = __builtin_amdgcn_wmma_f32_16x16x32_bf16(
        false, afr[1].v, false, bfr.v, (short)0, acc[1][s], false, false);
  }
}

// ---------------------------------------------------------------------------
// Kernel 0: one-time fp32 -> bf16 weight conversion (pairs -> dwords).
// ---------------------------------------------------------------------------
__global__ __launch_bounds__(256) void cvt_w_bf16(
    const float* __restrict__ src, unsigned* __restrict__ dst, int npairs) {
  const int i = blockIdx.x * 256 + threadIdx.x;
  if (i < npairs) {
    const float2 v = ((const float2*)src)[i];
    dst[i] = pack2(v.x, v.y);
  }
}

// ---------------------------------------------------------------------------
// Kernel 1: k/v projection. Rows 1..1024 of bf16 w_qkv -> kbuf / vbuf (+bias).
// grid (PN/BT_N=32, 1024/BT_M=8, 16), block 256.
// ---------------------------------------------------------------------------
__global__ __launch_bounds__(256) void qkv_gemm(
    const float* __restrict__ x, const unsigned short* __restrict__ wbf,
    const float* __restrict__ b_qkv,
    unsigned short* __restrict__ kbuf, unsigned short* __restrict__ vbuf) {
  __shared__ unsigned short Ws[BT_M * LSTR];
  __shared__ unsigned short Xs[BT_N * LSTR];
  const int tid = threadIdx.x;
  const int lane = tid & 31, wid = tid >> 5;
  const int waveM = wid >> 1, waveN = wid & 1;
  const int b = blockIdx.z;
  const int m_base = blockIdx.y * BT_M;
  const int n_base = blockIdx.x * BT_N;
  const float* xg = x + (size_t)b * (CCH * PN);

  const int wm = tid >> 1, wk = (tid & 1) * 16;       // W stage: 2 thr / row
  const int r2 = tid >> 4, nb = (tid & 15) * 8;       // X stage: k-pair, n-chunk

  v8f acc[2][4] = {};

  for (int kc = 0; kc < CCH; kc += KC) {
    {  // stage W tile (already bf16): one b128 load, two b64 LDS stores
      const uint4 wv =
          *(const uint4*)(wbf + (size_t)(1 + m_base + wm) * CCH + kc + wk);
      unsigned* dst = (unsigned*)&Ws[wm * LSTR + wk];
      *(uint2*)&dst[0] = make_uint2(wv.x, wv.y);
      *(uint2*)&dst[2] = make_uint2(wv.z, wv.w);
    }
    {  // stage X tile transposed: Xs[n][k], packing K-pairs into dwords
      const float* s0 = xg + (size_t)(kc + 2 * r2) * PN + n_base + nb;
      const float* s1 = s0 + PN;
      float a0[8], a1[8];
      *(float4*)&a0[0] = *(const float4*)(s0); *(float4*)&a0[4] = *(const float4*)(s0 + 4);
      *(float4*)&a1[0] = *(const float4*)(s1); *(float4*)&a1[4] = *(const float4*)(s1 + 4);
#pragma unroll
      for (int j = 0; j < 8; ++j)
        *(unsigned*)&Xs[(nb + j) * LSTR + 2 * r2] = pack2(a0[j], a1[j]);
    }
    __syncthreads();
    mma_chunk(Ws, Xs, waveM, waveN, lane, acc);
    __syncthreads();
  }

  const size_t bo = (size_t)b * (CCH * PN);
  const int lm = lane & 15;
  const int hrow = (lane >> 4) * 8;
#pragma unroll
  for (int a = 0; a < 2; ++a) {
#pragma unroll
    for (int g = 0; g < 8; ++g) {
      const int m = m_base + waveM * 32 + a * 16 + g + hrow;   // 0..1023
      const float bv = b_qkv[1 + m];
      unsigned short* dst = (m < CCH) ? (kbuf + bo + (size_t)m * PN)
                                      : (vbuf + bo + (size_t)(m - CCH) * PN);
#pragma unroll
      for (int s = 0; s < 4; ++s) {
        const int col = n_base + waveN * 64 + s * 16 + lm;
        dst[col] = f2bf(acc[a][s][g] + bv);
      }
    }
  }
}

// ---------------------------------------------------------------------------
// Kernel 2: query row matvec (row 0 of fp32 w_qkv). grid (16,16) block 256.
// ---------------------------------------------------------------------------
__global__ __launch_bounds__(256) void q_proj(
    const float* __restrict__ x, const float* __restrict__ w_qkv,
    const float* __restrict__ b_qkv, float* __restrict__ qbuf) {
  const int pn = blockIdx.x * 256 + threadIdx.x;
  const int b = blockIdx.y;
  const float* xp = x + (size_t)b * (CCH * PN) + pn;
  float s = b_qkv[0];
#pragma unroll 8
  for (int c = 0; c < CCH; ++c) s = fmaf(w_qkv[c], xp[(size_t)c * PN], s);
  qbuf[b * PN + pn] = s;
}

// ---------------------------------------------------------------------------
// Kernel 3: softmax over N=1024 per (b,p) row. grid 64, block 256 (4 elem/thr).
// ---------------------------------------------------------------------------
__global__ __launch_bounds__(256) void softmax_rows(
    const float* __restrict__ qbuf, float* __restrict__ scores) {
  __shared__ float red[8];
  __shared__ float stat;
  const int tid = threadIdx.x, lane = tid & 31, wid = tid >> 5;
  const float4 v = ((const float4*)(qbuf + (size_t)blockIdx.x * 1024))[tid];
  float m = fmaxf(fmaxf(v.x, v.y), fmaxf(v.z, v.w));
#pragma unroll
  for (int o = 16; o; o >>= 1) m = fmaxf(m, __shfl_xor(m, o, 32));
  if (lane == 0) red[wid] = m;
  __syncthreads();
  if (tid == 0) {
    float t = red[0];
    for (int i = 1; i < 8; ++i) t = fmaxf(t, red[i]);
    stat = t;
  }
  __syncthreads();
  const float M = stat;
  float4 e;
  e.x = expf(v.x - M); e.y = expf(v.y - M);
  e.z = expf(v.z - M); e.w = expf(v.w - M);
  float s = e.x + e.y + e.z + e.w;
#pragma unroll
  for (int o = 16; o; o >>= 1) s += __shfl_xor(s, o, 32);
  if (lane == 0) red[wid] = s;
  __syncthreads();
  if (tid == 0) {
    float t = 0.f;
    for (int i = 0; i < 8; ++i) t += red[i];
    stat = 1.0f / t;
  }
  __syncthreads();
  const float inv = stat;
  e.x *= inv; e.y *= inv; e.z *= inv; e.w *= inv;
  ((float4*)(scores + (size_t)blockIdx.x * 1024))[tid] = e;
}

// ---------------------------------------------------------------------------
// Kernel 4: context vector cv[b][c][p] = sum_n k*scores. One wave per (b,c,p).
// Lanes cover bf16 pairs (dword loads). grid 4096, block 256 (8 waves).
// ---------------------------------------------------------------------------
__global__ __launch_bounds__(256) void cv_reduce(
    const unsigned short* __restrict__ kbuf, const float* __restrict__ scores,
    float* __restrict__ cv) {
  const int lane = threadIdx.x & 31;
  const int gw = blockIdx.x * 8 + (threadIdx.x >> 5);
  const int b = gw >> 11;
  const int rem = gw & 2047;
  const int c = rem >> 2;
  const int p = rem & 3;
  const unsigned* kp =
      (const unsigned*)(kbuf + (size_t)(b * CCH + c) * PN + p * 1024);
  const float* sp = scores + (b * 4 + p) * 1024;
  float acc = 0.f;
#pragma unroll 4
  for (int i = lane; i < 512; i += 32) {       // i indexes bf16 pairs
    const unsigned kk = kp[i];
    acc = fmaf(bf2f((unsigned short)(kk & 0xFFFFu)), sp[2 * i], acc);
    acc = fmaf(bf2f((unsigned short)(kk >> 16)),     sp[2 * i + 1], acc);
  }
#pragma unroll
  for (int o = 16; o; o >>= 1) acc += __shfl_xor(acc, o, 32);
  if (lane == 0) cv[(b * CCH + c) * 4 + p] = acc;
}

// ---------------------------------------------------------------------------
// Kernel 5: out = w_out @ (relu(v)*cv) + b_out, relu/scale fused into staging.
// grid (32, 512/BT_M=4, 16), block 256.
// ---------------------------------------------------------------------------
__global__ __launch_bounds__(256) void out_gemm(
    const unsigned short* __restrict__ vbuf, const unsigned short* __restrict__ wbf,
    const float* __restrict__ b_out, const float* __restrict__ cv,
    float* __restrict__ out) {
  __shared__ unsigned short Ws[BT_M * LSTR];
  __shared__ unsigned short Xs[BT_N * LSTR];
  const int tid = threadIdx.x;
  const int lane = tid & 31, wid = tid >> 5;
  const int waveM = wid >> 1, waveN = wid & 1;
  const int b = blockIdx.z;
  const int m_base = blockIdx.y * BT_M;
  const int n_base = blockIdx.x * BT_N;
  const size_t bo = (size_t)b * (CCH * PN);

  const int wm = tid >> 1, wk = (tid & 1) * 16;
  const int r2 = tid >> 4, nb = (tid & 15) * 8;

  v8f acc[2][4] = {};

  for (int kc = 0; kc < CCH; kc += KC) {
    {  // stage w_out tile (already bf16)
      const uint4 wv =
          *(const uint4*)(wbf + (size_t)(m_base + wm) * CCH + kc + wk);
      unsigned* dst = (unsigned*)&Ws[wm * LSTR + wk];
      *(uint2*)&dst[0] = make_uint2(wv.x, wv.y);
      *(uint2*)&dst[2] = make_uint2(wv.z, wv.w);
    }
    {  // stage relu(v)*cv tile, transposed bf16
      const int c0 = kc + 2 * r2;
      const int p = (n_base + nb) >> 10;   // constant across the 8-wide chunk
      const float s0 = cv[(b * CCH + c0) * 4 + p];
      const float s1 = cv[(b * CCH + c0 + 1) * 4 + p];
      const unsigned short* v0 = vbuf + bo + (size_t)c0 * PN + n_base + nb;
      const unsigned short* v1 = v0 + PN;
      union { uint4 q; unsigned short h[8]; } u0, u1;
      u0.q = *(const uint4*)v0;
      u1.q = *(const uint4*)v1;
#pragma unroll
      for (int j = 0; j < 8; ++j) {
        const float f0 = fmaxf(bf2f(u0.h[j]), 0.f) * s0;
        const float f1 = fmaxf(bf2f(u1.h[j]), 0.f) * s1;
        *(unsigned*)&Xs[(nb + j) * LSTR + 2 * r2] = pack2(f0, f1);
      }
    }
    __syncthreads();
    mma_chunk(Ws, Xs, waveM, waveN, lane, acc);
    __syncthreads();
  }

  const int lm = lane & 15;
  const int hrow = (lane >> 4) * 8;
#pragma unroll
  for (int a = 0; a < 2; ++a) {
#pragma unroll
    for (int g = 0; g < 8; ++g) {
      const int m = m_base + waveM * 32 + a * 16 + g + hrow;
      const float bv = b_out[m];
      float* dst = out + bo + (size_t)m * PN;
#pragma unroll
      for (int s = 0; s < 4; ++s) {
        const int col = n_base + waveN * 64 + s * 16 + lm;
        dst[col] = acc[a][s][g] + bv;
      }
    }
  }
}

// ---------------------------------------------------------------------------
// Workspace layout (bytes, all 16B-aligned):
//   kbuf    bf16 [16][512][4096]  = 67,108,864
//   vbuf    bf16 [16][512][4096]  = 67,108,864
//   qbuf    f32  [16][4096]       =    262,144
//   scores  f32  [16][4][1024]    =    262,144
//   cv      f32  [16][512][4]     =    131,072
//   wqkv_bf bf16 [1025][512]      =  1,049,600
//   wout_bf bf16 [512][512]       =    524,288
// total ~136.4 MB
// ---------------------------------------------------------------------------
extern "C" void kernel_launch(void* const* d_in, const int* in_sizes, int n_in,
                              void* d_out, int out_size, void* d_ws, size_t ws_size,
                              hipStream_t stream) {
  const float* x     = (const float*)d_in[0];
  const float* w_qkv = (const float*)d_in[1];
  const float* b_qkv = (const float*)d_in[2];
  const float* w_out = (const float*)d_in[3];
  const float* b_out = (const float*)d_in[4];
  float* out = (float*)d_out;

  unsigned short* kbuf = (unsigned short*)d_ws;
  unsigned short* vbuf = kbuf + (size_t)NB * CCH * PN;
  float* qbuf   = (float*)(vbuf + (size_t)NB * CCH * PN);
  float* scores = qbuf + (size_t)NB * PN;
  float* cv     = scores + (size_t)NB * PN;
  unsigned short* wqkv_bf = (unsigned short*)(cv + (size_t)NB * CCH * 4);
  unsigned short* wout_bf = wqkv_bf + (size_t)(1 + 2 * CCH) * CCH;

  const int npq = (1 + 2 * CCH) * CCH / 2;   // 262,400 pairs
  const int npo = CCH * CCH / 2;             // 131,072 pairs
  cvt_w_bf16<<<(npq + 255) / 256, 256, 0, stream>>>(w_qkv, (unsigned*)wqkv_bf, npq);
  cvt_w_bf16<<<(npo + 255) / 256, 256, 0, stream>>>(w_out, (unsigned*)wout_bf, npo);

  qkv_gemm<<<dim3(PN / BT_N, 1024 / BT_M, NB), 256, 0, stream>>>(x, wqkv_bf, b_qkv, kbuf, vbuf);
  q_proj<<<dim3(PN / 256, NB), 256, 0, stream>>>(x, w_qkv, b_qkv, qbuf);
  softmax_rows<<<NB * 4, 256, 0, stream>>>(qbuf, scores);
  cv_reduce<<<(NB * CCH * 4) / 8, 256, 0, stream>>>(kbuf, scores, cv);
  out_gemm<<<dim3(PN / BT_N, CCH / BT_M, NB), 256, 0, stream>>>(vbuf, wout_bf, b_out, cv, out);
}
